// TransLayer_70798240907449
// MI455X (gfx1250) — matrix operand order repaired and checked by reading, not compiled, not run on hardware
//
#include <hip/hip_runtime.h>

typedef unsigned short u16;
typedef unsigned int   u32;
typedef unsigned long long u64;
typedef long long      i64;
typedef __attribute__((ext_vector_type(16))) __bf16 v16bf;
typedef __attribute__((ext_vector_type(8)))  float  v8f;
typedef __attribute__((ext_vector_type(2)))  float  v2f;
typedef __attribute__((ext_vector_type(4)))  unsigned int v4u;
typedef __attribute__((ext_vector_type(8)))  int    v8i;
typedef __attribute__((ext_vector_type(4)))  int    v4i;

constexpr int kB   = 4;
constexpr int kN   = 8192;
constexpr int kD   = 512;     // model dim == INNER
constexpr int kH   = 8;
constexpr int kDH  = 64;
constexpr int kLM  = 256;     // landmarks
constexpr int kBH  = kB * kH;     // 32
constexpr int kTOT = kB * kN;     // 32768
constexpr int kKER = 33;

__device__ __forceinline__ float bf2f(u16 s) { return __uint_as_float(((u32)s) << 16); }
__device__ __forceinline__ u16 f2bf(float f) {
  u32 u = __float_as_uint(f);
  u32 r = u + 0x7FFFu + ((u >> 16) & 1u);
  return (u16)(r >> 16);
}

union Frag16 { v16bf v; u32 u[8]; };
union FragC  { v8f  v; float f[8]; };

// A fragment (16x32 bf16), per ISA 7.12.2: lane<16 -> K {0..7,16..23}, lane>=16 -> +8
__device__ __forceinline__ v16bf ldsA(const u16* tile, int stride, int lane) {
  Frag16 f;
  const u16* rp = tile + (lane & 15) * stride;
  int kb = ((lane >> 4) << 3);
#pragma unroll
  for (int p = 0; p < 8; ++p) {
    int k = kb + ((p < 4) ? (p << 1) : (16 + ((p - 4) << 1)));
    f.u[p] = *(const u32*)(rp + k);
  }
  return f.v;
}
// B fragment (32x16 bf16), stored transposed in LDS as [n][k]; lane<16 -> K {0..7,16..23}, lane>=16 -> +8
__device__ __forceinline__ v16bf ldsB(const u16* tile, int stride, int lane) {
  Frag16 f;
  const u16* rp = tile + (lane & 15) * stride;
  int kh = ((lane >> 4) << 3);
#pragma unroll
  for (int p = 0; p < 8; ++p) f.u[p] = *(const u32*)(rp + kh + ((p < 4) ? (p << 1) : (16 + ((p - 4) << 1))));
  return f.v;
}
// f32 fragment (16x4 A / 4x16 B^T), ISA: lanes 0-15 -> K=k+{0,1}, lanes 16-31 -> K=k+{2,3}
__device__ __forceinline__ v2f ldsF2(const float* tile, int stride, int lane, int k) {
  const float* rp = tile + (lane & 15) * stride + k + ((lane >> 4) << 1);
  v2f r; r.x = rp[0]; r.y = rp[1];
  return r;
}

// ---------------- generic batched bf16 WMMA GEMM ----------------
// C[M,N] = scale * A[M,K] @ B(K,N) (+bias[n]) (+resid);  bt: B given as B^T [N,K]
// block tile 128(M) x 64(N), 4 waves, each wave owns 32 rows (two 16-row WMMA tiles).
// A tile staged by the Tensor Data Mover (TDM) with LDS padding to a 34-u16 row pitch.
// batch z: off = (z>>3)*sHi + (z&7)*sLo for each of A/B/C.
__global__ __launch_bounds__(128)
void gemm_bf16(const u16* __restrict__ A, const u16* __restrict__ B, void* __restrict__ Cout,
               int N, int K, int lda, int ldb, int ldc,
               i64 saLo, i64 saHi, i64 sbLo, i64 sbHi, i64 scLo, i64 scHi,
               int bt, int obf16, const float* __restrict__ bias,
               const float* __restrict__ resid, float scale)
{
  const int z = blockIdx.z, zlo = z & 7, zhi = z >> 3;
  const u16* Ab = A + zhi * saHi + zlo * saLo;
  const u16* Bb = B + zhi * sbHi + zlo * sbLo;
  const i64  cOff = zhi * scHi + zlo * scLo;

  const int m0 = blockIdx.y * 128;
  const int n0 = blockIdx.x * 64;

  __shared__ __align__(16) u16 As[128][34];
  __shared__ __align__(16) u16 Bs[64][34];

  const int tid = threadIdx.x, wave = tid >> 5, lane = tid & 31;
  const u32 ldsAbase = (u32)(uintptr_t)&As[0][0];   // LDS aperture: byte offset in addr[31:0]

  FragC c[2][4];
#pragma unroll
  for (int rt = 0; rt < 2; ++rt)
#pragma unroll
    for (int t = 0; t < 4; ++t)
#pragma unroll
      for (int v = 0; v < 8; ++v) c[rt][t].f[v] = 0.f;

  for (int k0 = 0; k0 < K; k0 += 32) {
    // ---- A tile: 128x32 bf16 via TENSOR_LOAD_TO_LDS (pad 1 dword per 16-dword row) ----
    if (tid < 32) {
      u64 ga = (u64)(uintptr_t)(Ab + (i64)m0 * lda + k0);
      v4u g0 = { 1u,                                   // count=1, user descriptor
                 ldsAbase,                             // lds_addr
                 (u32)ga,                              // global_addr[31:0]
                 ((u32)(ga >> 32) & 0x01FFFFFFu) | 0x80000000u };  // addr[56:32] | type=2
      v8i g1 = { (int)0x00D10000u,                     // data_size=2B, pad_en, interval=16dw, amount=1dw
                 (int)(32u << 16),                     // tensor_dim0 = 32
                 (int)(128u << 16),                    // tensor_dim1 = 128
                 (int)(32u << 16),                     // tile_dim0 = 32
                 128,                                  // tile_dim1 = 128
                 (int)(u32)lda,                        // tensor_dim0_stride
                 0, 0 };
      v4i gz4 = { 0, 0, 0, 0 };
      v8i gz8 = { 0, 0, 0, 0, 0, 0, 0, 0 };
      __builtin_amdgcn_tensor_load_to_lds(g0, g1, gz4, gz4, gz8, 0);
      __builtin_amdgcn_s_wait_tensorcnt(0);
    }
    // ---- B tile ----
    if (k0 + 32 < K) {   // prefetch next chunk of B (global_prefetch_b8)
      if (bt) { if (tid < 64) __builtin_prefetch(Bb + (i64)(n0 + tid) * ldb + (k0 + 32), 0, 0); }
      else    { if (tid < 32) __builtin_prefetch(Bb + (i64)(k0 + 32 + tid) * ldb + n0, 0, 0); }
    }
    if (bt) {
      for (int i = tid; i < 64 * 16; i += 128) {
        int n = i >> 4, kk = (i & 15) << 1;
        *(u32*)&Bs[n][kk] = *(const u32*)(Bb + (i64)(n0 + n) * ldb + (k0 + kk));
      }
    } else {
      for (int i = tid; i < 64 * 32; i += 128) {
        int n = i & 63, kk = i >> 6;
        Bs[n][kk] = Bb[(i64)(k0 + kk) * ldb + (n0 + n)];
      }
    }
    __syncthreads();
    v16bf a0 = ldsA(&As[wave * 32][0], 34, lane);
    v16bf a1 = ldsA(&As[wave * 32 + 16][0], 34, lane);
#pragma unroll
    for (int nt = 0; nt < 4; ++nt) {
      v16bf b = ldsB(&Bs[nt * 16][0], 34, lane);
      c[0][nt].v = __builtin_amdgcn_wmma_f32_16x16x32_bf16(false, a0, false, b, (short)0,
                                                           c[0][nt].v, false, false);
      c[1][nt].v = __builtin_amdgcn_wmma_f32_16x16x32_bf16(false, a1, false, b, (short)0,
                                                           c[1][nt].v, false, false);
    }
    __syncthreads();
  }

  const int cb = lane & 15;
#pragma unroll
  for (int rt = 0; rt < 2; ++rt) {
    const int rbase = m0 + wave * 32 + rt * 16 + ((lane >> 4) << 3);
#pragma unroll
    for (int nt = 0; nt < 4; ++nt) {
      int col = n0 + nt * 16 + cb;
#pragma unroll
      for (int v = 0; v < 8; ++v) {
        float val = c[rt][nt].f[v] * scale;
        if (bias) val += bias[col];
        i64 addr = cOff + (i64)(rbase + v) * ldc + col;
        if (resid) val += resid[addr];
        if (obf16) ((u16*)Cout)[addr] = f2bf(val);
        else       ((float*)Cout)[addr] = val;
      }
    }
  }
}

// ---------------- 256x256 batched matmul, full f32 via V_WMMA_F32_16X16X4_F32 ----------------
__global__ __launch_bounds__(256)
void mm256(const float* __restrict__ A, const float* __restrict__ B,
           float* __restrict__ C, float scale)
{
  const int z = blockIdx.z;
  const float* Ab = A + (i64)z * 65536;
  const float* Bb = B + (i64)z * 65536;
  float* Cb = C + (i64)z * 65536;
  const int m0 = blockIdx.y * 128, n0 = blockIdx.x * 128;

  __shared__ __align__(16) float As[128][36];
  __shared__ __align__(16) float Bs[128][36];   // B stored transposed: [n][k]

  const int tid = threadIdx.x, wave = tid >> 5, lane = tid & 31;

  FragC c[8];
#pragma unroll
  for (int t = 0; t < 8; ++t)
#pragma unroll
    for (int v = 0; v < 8; ++v) c[t].f[v] = 0.f;

  for (int k0 = 0; k0 < 256; k0 += 32) {
    for (int i = tid; i < 128 * 32; i += 256) {
      int r = i >> 5, kk = i & 31;
      As[r][kk] = Ab[(i64)(m0 + r) * 256 + k0 + kk];
    }
    for (int i = tid; i < 128 * 32; i += 256) {
      int n = i & 127, kk = i >> 7;
      Bs[n][kk] = Bb[(i64)(k0 + kk) * 256 + n0 + n];
    }
    __syncthreads();
    const float* arow = &As[wave * 16][0];
#pragma unroll
    for (int ks = 0; ks < 8; ++ks) {
      v2f a = ldsF2(arow, 36, lane, ks * 4);
#pragma unroll
      for (int nt = 0; nt < 8; ++nt) {
        v2f b = ldsF2(&Bs[nt * 16][0], 36, lane, ks * 4);
        c[nt].v = __builtin_amdgcn_wmma_f32_16x16x4_f32(false, a, false, b, (short)0,
                                                        c[nt].v, false, false);
      }
    }
    __syncthreads();
  }

  const int rbase = m0 + wave * 16 + ((lane >> 4) << 3);
  const int cb = lane & 15;
#pragma unroll
  for (int nt = 0; nt < 8; ++nt) {
    int col = n0 + nt * 16 + cb;
#pragma unroll
    for (int v = 0; v < 8; ++v)
      Cb[(i64)(rbase + v) * 256 + col] = scale * c[nt].f[v];
  }
}

// ---------------- elementwise / reduction kernels ----------------
__global__ __launch_bounds__(256)
void layernorm_cast(const float* __restrict__ x, const float* __restrict__ gamma,
                    const float* __restrict__ beta, u16* __restrict__ out)
{
  const int row = blockIdx.x;
  const float* xr = x + (i64)row * kD;
  const int t = threadIdx.x;
  float a = xr[t], b = xr[t + 256];
  __shared__ float s1[256], s2[256];
  s1[t] = a + b; s2[t] = a * a + b * b;
  __syncthreads();
  for (int s = 128; s > 0; s >>= 1) {
    if (t < s) { s1[t] += s1[t + s]; s2[t] += s2[t + s]; }
    __syncthreads();
  }
  float mu = s1[0] * (1.f / kD);
  float var = s2[0] * (1.f / kD) - mu * mu;
  float rs = rsqrtf(var + 1e-5f);
  u16* orow = out + (i64)row * kD;
  orow[t]       = f2bf((a - mu) * rs * gamma[t] + beta[t]);
  orow[t + 256] = f2bf((b - mu) * rs * gamma[t + 256] + beta[t + 256]);
}

__global__ void cast_wqkv(const float* __restrict__ w, u16* __restrict__ o) {
  int i = blockIdx.x * 256 + threadIdx.x;
  if (i >= kD * 3 * kD) return;
  int col = i % (3 * kD);
  float s = (col < kD) ? 0.125f : 1.f;   // fold q scale DIM_HEAD^-0.5
  o[i] = f2bf(w[i] * s);
}

__global__ void cast_bf16(const float* __restrict__ s, u16* __restrict__ o, int n) {
  int i = blockIdx.x * 256 + threadIdx.x;
  if (i < n) o[i] = f2bf(s[i]);
}

__global__ void landmarks(const u16* __restrict__ qkv, u16* __restrict__ ql, u16* __restrict__ kl) {
  int idx = blockIdx.x * 256 + threadIdx.x;
  if (idx >= kBH * kLM * kDH) return;
  int d = idx & 63, lm = (idx >> 6) & 255, z = idx >> 14;
  int b = z >> 3, h = z & 7;
  i64 base = ((i64)(b * kN + lm * 32)) * (3 * kD) + h * kDH + d;
  float sq = 0.f, sk = 0.f;
  for (int r = 0; r < 32; ++r) {
    i64 a = base + (i64)r * (3 * kD);
    sq += bf2f(qkv[a]);
    sk += bf2f(qkv[a + kD]);
  }
  ql[idx] = f2bf(sq * (1.f / 32.f));
  kl[idx] = f2bf(sk * (1.f / 32.f));
}

__global__ __launch_bounds__(256)
void softmax_rows(const float* __restrict__ src, float* __restrict__ dstF,
                  u16* __restrict__ dstB, int L)
{
  const i64 row = blockIdx.x;
  const float* r = src + row * (i64)L;
  const int t = threadIdx.x;
  __shared__ float red[256];
  float mx = -3.0e38f;
  for (int i = t; i < L; i += 256) mx = fmaxf(mx, r[i]);
  red[t] = mx; __syncthreads();
  for (int s = 128; s > 0; s >>= 1) { if (t < s) red[t] = fmaxf(red[t], red[t + s]); __syncthreads(); }
  const float m = red[0]; __syncthreads();
  float sm = 0.f;
  for (int i = t; i < L; i += 256) sm += __expf(r[i] - m);
  red[t] = sm; __syncthreads();
  for (int s = 128; s > 0; s >>= 1) { if (t < s) red[t] += red[t + s]; __syncthreads(); }
  const float inv = 1.f / red[0];
  for (int i = t; i < L; i += 256) {
    float p = __expf(r[i] - m) * inv;
    if (dstF) dstF[row * (i64)L + i] = p;
    if (dstB) dstB[row * (i64)L + i] = f2bf(p);
  }
}

__global__ void zero2(u32* p) { if (threadIdx.x < 2) p[threadIdx.x] = 0u; }

__global__ __launch_bounds__(256)
void colrowmax(const float* __restrict__ a2, u32* __restrict__ sc) {
  int z = blockIdx.x;
  const float* A = a2 + (i64)z * 65536;
  int t = threadIdx.x;
  float rs = 0.f, cs = 0.f;
  for (int i = 0; i < 256; ++i) { rs += fabsf(A[t * 256 + i]); cs += fabsf(A[i * 256 + t]); }
  __shared__ float r1[256], r2[256];
  r1[t] = rs; r2[t] = cs; __syncthreads();
  for (int s = 128; s > 0; s >>= 1) {
    if (t < s) { r1[t] = fmaxf(r1[t], r1[t + s]); r2[t] = fmaxf(r2[t], r2[t + s]); }
    __syncthreads();
  }
  if (t == 0) {
    atomicMax(&sc[0], __float_as_uint(r1[0]));   // positive floats: uint order == float order
    atomicMax(&sc[1], __float_as_uint(r2[0]));
  }
}

__global__ void pinv_init(const float* __restrict__ a2, const u32* __restrict__ sc,
                          float* __restrict__ z) {
  int i = blockIdx.x * 256 + threadIdx.x;
  if (i >= kBH * 65536) return;
  int zb = i >> 16, r = (i >> 8) & 255, c = i & 255;
  float denom = __uint_as_float(sc[0]) * __uint_as_float(sc[1]);
  z[i] = a2[((i64)zb << 16) + (c << 8) + r] / denom;  // transpose / denom
}

__global__ void eyesub(const float* __restrict__ in, float* __restrict__ out, float alpha) {
  int i = blockIdx.x * 256 + threadIdx.x;
  if (i >= kBH * 65536) return;
  int r = (i >> 8) & 255, c = i & 255;
  out[i] = ((r == c) ? alpha : 0.f) - in[i];
}

__global__ void convadd(float* __restrict__ oh, const u16* __restrict__ qkv,
                        const float* __restrict__ cw) {
  int i = blockIdx.x * 256 + threadIdx.x;
  if (i >= kBH * kN * kDH) return;
  int d = i & 63, n = (i >> 6) & (kN - 1), z = i >> 19;
  int b = z >> 3, h = z & 7;
  float acc = 0.f;
  i64 base = (i64)b * kN * (3 * kD) + 2 * kD + h * kDH + d;  // v slice
  for (int t = 0; t < kKER; ++t) {
    int nn = n + t - kKER / 2;
    if (nn >= 0 && nn < kN) acc += cw[h * kKER + t] * bf2f(qkv[base + (i64)nn * (3 * kD)]);
  }
  oh[i] += acc;
}

__global__ void pack_heads(const float* __restrict__ oh, u16* __restrict__ ab) {
  int i = blockIdx.x * 256 + threadIdx.x;
  if (i >= kBH * kN * kDH) return;
  int d = i & 63, n = (i >> 6) & (kN - 1), z = i >> 19;
  int b = z >> 3, h = z & 7;
  ab[((i64)(b * kN + n)) * kD + h * kDH + d] = f2bf(oh[i]);
}

// ---------------- host orchestration ----------------
extern "C" void kernel_launch(void* const* d_in, const int* in_sizes, int n_in,
                              void* d_out, int out_size, void* d_ws, size_t ws_size,
                              hipStream_t stream)
{
  (void)in_sizes; (void)n_in; (void)out_size; (void)ws_size;
  const float* x      = (const float*)d_in[0];
  const float* gamma  = (const float*)d_in[1];
  const float* beta   = (const float*)d_in[2];
  const float* w_qkv  = (const float*)d_in[3];
  const float* w_out  = (const float*)d_in[4];
  const float* b_out  = (const float*)d_in[5];
  const float* conv_w = (const float*)d_in[6];
  float* out = (float*)d_out;

  char* ws = (char*)d_ws;
  size_t off = 0;
  auto alloc = [&](size_t bytes) -> void* {
    size_t cur = (off + 255) & ~(size_t)255;
    off = cur + bytes;
    return (void*)(ws + cur);
  };

  u16*   wqkvB  = (u16*)alloc((size_t)kD * 3 * kD * 2);
  u16*   woutB  = (u16*)alloc((size_t)kD * kD * 2);
  u16*   xB     = (u16*)alloc((size_t)kTOT * kD * 2);
  u16*   qkvB   = (u16*)alloc((size_t)kTOT * 3 * kD * 2);
  u16*   qlB    = (u16*)alloc((size_t)kBH * kLM * kDH * 2);
  u16*   klB    = (u16*)alloc((size_t)kBH * kLM * kDH * 2);
  float* attn2  = (float*)alloc((size_t)kBH * kLM * kLM * 4);
  float* zA     = (float*)alloc((size_t)kBH * kLM * kLM * 4);
  float* zBv    = (float*)alloc((size_t)kBH * kLM * kLM * 4);
  float* xz     = (float*)alloc((size_t)kBH * kLM * kLM * 4);
  float* sB     = (float*)alloc((size_t)kBH * kLM * kLM * 4);
  float* tB     = (float*)alloc((size_t)kBH * kLM * kLM * 4);
  float* simBig = (float*)alloc((size_t)kBH * kLM * kN * 4);   // reused: sim3 then sim1
  u16*   pBig   = (u16*)alloc((size_t)kBH * kLM * kN * 2);     // reused: P3 then P1
  float* W2     = (float*)alloc((size_t)kBH * kLM * kDH * 4);
  u16*   W2b    = (u16*)alloc((size_t)kBH * kLM * kDH * 2);
  u16*   zInvB  = (u16*)alloc((size_t)kBH * kLM * kLM * 2);
  u16*   W3b    = (u16*)alloc((size_t)kBH * kLM * kDH * 2);
  float* ohead  = (float*)alloc((size_t)kBH * kN * kDH * 4);
  u16*   aB     = (u16*)alloc((size_t)kTOT * kD * 2);
  u32*   scal   = (u32*)alloc(256);

  cast_wqkv<<<dim3((kD * 3 * kD + 255) / 256), 256, 0, stream>>>(w_qkv, wqkvB);
  cast_bf16<<<dim3((kD * kD + 255) / 256), 256, 0, stream>>>(w_out, woutB, kD * kD);
  layernorm_cast<<<dim3(kTOT), 256, 0, stream>>>(x, gamma, beta, xB);

  // qkv = LN(x) @ w_qkv  (bf16 out, q pre-scaled via weights)
  gemm_bf16<<<dim3(3 * kD / 64, kTOT / 128, 1), 128, 0, stream>>>(
      xB, wqkvB, qkvB, 3 * kD, kD, kD, 3 * kD, 3 * kD,
      0, 0, 0, 0, 0, 0, 0, 1, nullptr, nullptr, 1.f);

  landmarks<<<dim3((kBH * kLM * kDH + 255) / 256), 256, 0, stream>>>(qkvB, qlB, klB);

  // sim2 = qL @ kL^T
  gemm_bf16<<<dim3(kLM / 64, kLM / 128, kBH), 128, 0, stream>>>(
      qlB, klB, attn2, kLM, kDH, kDH, kDH, kLM,
      (i64)kLM * kDH, (i64)8 * kLM * kDH, (i64)kLM * kDH, (i64)8 * kLM * kDH,
      (i64)kLM * kLM, (i64)8 * kLM * kLM, 1, 0, nullptr, nullptr, 1.f);
  softmax_rows<<<dim3(kBH * kLM), 256, 0, stream>>>(attn2, attn2, nullptr, kLM);

  // Moore-Penrose pinv (Newton-Schulz, global denom like reference)
  zero2<<<1, 32, 0, stream>>>(scal);
  colrowmax<<<dim3(kBH), 256, 0, stream>>>(attn2, scal);
  pinv_init<<<dim3(kBH * 65536 / 256), 256, 0, stream>>>(attn2, scal, zA);

  float* zc = zA; float* za = zBv;
  dim3 gm(2, 2, kBH);
  int eb = kBH * 65536 / 256;
  for (int it = 0; it < 6; ++it) {
    mm256<<<gm, 256, 0, stream>>>(attn2, zc, xz, 1.f);
    eyesub<<<eb, 256, 0, stream>>>(xz, sB, 7.f);
    mm256<<<gm, 256, 0, stream>>>(xz, sB, tB, 1.f);
    eyesub<<<eb, 256, 0, stream>>>(tB, sB, 15.f);
    mm256<<<gm, 256, 0, stream>>>(xz, sB, tB, 1.f);
    eyesub<<<eb, 256, 0, stream>>>(tB, sB, 13.f);
    mm256<<<gm, 256, 0, stream>>>(zc, sB, za, 0.25f);
    float* tmp = zc; zc = za; za = tmp;
  }

  // sim3 = qL @ k^T ; softmax -> P3 (bf16)
  gemm_bf16<<<dim3(kN / 64, kLM / 128, kBH), 128, 0, stream>>>(
      qlB, qkvB + kD, simBig, kN, kDH, kDH, 3 * kD, kN,
      (i64)kLM * kDH, (i64)8 * kLM * kDH, (i64)kDH, (i64)kN * 3 * kD,
      (i64)kLM * kN, (i64)8 * kLM * kN, 1, 0, nullptr, nullptr, 1.f);
  softmax_rows<<<dim3(kBH * kLM), 256, 0, stream>>>(simBig, nullptr, pBig, kN);

  // W2 = P3 @ v
  gemm_bf16<<<dim3(kDH / 64, kLM / 128, kBH), 128, 0, stream>>>(
      pBig, qkvB + 2 * kD, W2, kDH, kN, kN, 3 * kD, kDH,
      (i64)kLM * kN, (i64)8 * kLM * kN, (i64)kDH, (i64)kN * 3 * kD,
      (i64)kLM * kDH, (i64)8 * kLM * kDH, 0, 0, nullptr, nullptr, 1.f);

  cast_bf16<<<dim3((kBH * kLM * kLM + 255) / 256), 256, 0, stream>>>(zc, zInvB, kBH * kLM * kLM);
  cast_bf16<<<dim3((kBH * kLM * kDH + 255) / 256), 256, 0, stream>>>(W2, W2b, kBH * kLM * kDH);

  // W3 = attn2_inv @ W2  (bf16 out)
  gemm_bf16<<<dim3(kDH / 64, kLM / 128, kBH), 128, 0, stream>>>(
      zInvB, W2b, W3b, kDH, kLM, kLM, kDH, kDH,
      (i64)kLM * kLM, (i64)8 * kLM * kLM, (i64)kLM * kDH, (i64)8 * kLM * kDH,
      (i64)kLM * kDH, (i64)8 * kLM * kDH, 0, 1, nullptr, nullptr, 1.f);

  // sim1 = q @ kL^T ; softmax -> P1 (bf16)
  gemm_bf16<<<dim3(kLM / 64, kN / 128, kBH), 128, 0, stream>>>(
      qkvB, klB, simBig, kLM, kDH, 3 * kD, kDH, kLM,
      (i64)kDH, (i64)kN * 3 * kD, (i64)kLM * kDH, (i64)8 * kLM * kDH,
      (i64)kN * kLM, (i64)8 * kN * kLM, 1, 0, nullptr, nullptr, 1.f);
  softmax_rows<<<dim3(kBH * kN), 256, 0, stream>>>(simBig, nullptr, pBig, kLM);

  // ohead = P1 @ W3
  gemm_bf16<<<dim3(kDH / 64, kN / 128, kBH), 128, 0, stream>>>(
      pBig, W3b, ohead, kDH, kLM, kLM, kDH, kDH,
      (i64)kN * kLM, (i64)8 * kN * kLM, (i64)kLM * kDH, (i64)8 * kLM * kDH,
      (i64)kN * kDH, (i64)8 * kN * kDH, 0, 0, nullptr, nullptr, 1.f);

  convadd<<<dim3(kBH * kN * kDH / 256), 256, 0, stream>>>(ohead, qkvB, conv_w);
  pack_heads<<<dim3(kBH * kN * kDH / 256), 256, 0, stream>>>(ohead, aB);

  // out = x + a @ w_out + b_out
  gemm_bf16<<<dim3(kD / 64, kTOT / 128, 1), 128, 0, stream>>>(
      aB, woutB, out, kD, kD, kD, kD, kD,
      0, 0, 0, 0, 0, 0, 0, 0, b_out, x, 1.f);
}
